// GCN_11793980195193
// MI455X (gfx1250) — compile-verified, matrix operand order checked by reading, not run on hardware
//
#include <hip/hip_runtime.h>

typedef _Float16 half_t;
typedef __attribute__((ext_vector_type(16))) _Float16 v16h;
typedef __attribute__((ext_vector_type(8)))  float    v8f;

// ---------------- small elementwise / degree kernels ----------------

__global__ void fill1_kernel(float* __restrict__ p, int n) {
  int i = blockIdx.x * blockDim.x + threadIdx.x;
  if (i < n) p[i] = 1.0f;   // self-loop contributes 1 to degree
}

__global__ void deg_accum_kernel(const int* __restrict__ dst, float* deg, int E) {
  int e = blockIdx.x * blockDim.x + threadIdx.x;
  if (e < E) atomicAdd(&deg[dst[e]], 1.0f);
}

__global__ void dinv_kernel(const float* __restrict__ deg, float* __restrict__ dinv, int n) {
  int i = blockIdx.x * blockDim.x + threadIdx.x;
  if (i < n) dinv[i] = rsqrtf(deg[i]);
}

// out[i*F+f] = bias[f] + H[i*Hs+f] * dinv[i]^2   (bias + self-loop message)
__global__ void node_init_kernel(const float* __restrict__ H, const float* __restrict__ dinv,
                                 const float* __restrict__ bias, float* __restrict__ out,
                                 int n, int F, int Hs) {
  int t = blockIdx.x * blockDim.x + threadIdx.x;
  if (t >= n * F) return;
  int i = t / F, f = t - i * F;
  float di = dinv[i];
  out[(size_t)i * F + f] = bias[f] + H[(size_t)i * Hs + f] * di * di;
}

// Layer-1 scatter, F=64: 16 threads/edge, float4 gather + 4 f32 atomics each.
__global__ void scatter64_kernel(const float* __restrict__ H, const int* __restrict__ src,
                                 const int* __restrict__ dst, const float* __restrict__ dinv,
                                 float* out, int E) {
  int t = blockIdx.x * blockDim.x + threadIdx.x;
  if (t >= E * 16) return;              // E*16 = 25.6M, fits int
  int e = t >> 4, q = (t & 15) << 2;
  int s = src[e], d = dst[e];
  float nrm = dinv[s] * dinv[d];
  const float4 h = *(const float4*)&H[(size_t)s * 64 + q];
  float* o = &out[(size_t)d * 64 + q];
  atomicAdd(o + 0, h.x * nrm);
  atomicAdd(o + 1, h.y * nrm);
  atomicAdd(o + 2, h.z * nrm);
  atomicAdd(o + 3, h.w * nrm);
}

// Generic scatter (layer 2, F=41): one (edge, feature) per thread.
__global__ void scatter_kernel(const float* __restrict__ H, const int* __restrict__ src,
                               const int* __restrict__ dst, const float* __restrict__ dinv,
                               float* out, int E, int F, int Hs) {
  int t = blockIdx.x * blockDim.x + threadIdx.x;
  if (t >= E * F) return;               // E*41 = 65.6M, fits int
  int e = t / F, f = t - e * F;
  int s = src[e], d = dst[e];
  float nrm = dinv[s] * dinv[d];
  atomicAdd(&out[(size_t)d * F + f], H[(size_t)s * Hs + f] * nrm);
}

__global__ void relu_kernel(float* __restrict__ h, int total) {
  int i = blockIdx.x * blockDim.x + threadIdx.x;
  if (i < total) h[i] = fmaxf(h[i], 0.0f);
}

// ---------------- WMMA GEMM: C[M,NPAD] = A[M,K] x W[K,NREAL] (zero-padded to NPAD) ----
// One wave per 16-row M tile, all NPAD/16 column tiles.
// W is pre-swizzled into LDS in the exact per-lane B-fragment layout so each
// fragment is one contiguous 32B read per lane (ds_load_b128 x2).

template<int K, int NREAL, int NPAD>
__global__ __launch_bounds__(256) void gemm_wmma_kernel(
    const float* __restrict__ A, const float* __restrict__ W,
    float* __restrict__ C, int Mtiles, int M)
{
  constexpr int NT = NPAD / 16;    // column tiles
  constexpr int KC = K / 32;       // K chunks
  // Fragment-ordered: Wl[((kc*NT + nt)*32 + lane)*16 + j]
  __shared__ __align__(32) half_t Wl[KC * NT * 32 * 16];

  for (int i = threadIdx.x; i < KC * NT * 512; i += 256) {
    int j    = i & 15;
    int lane = (i >> 4) & 31;
    int nt   = (i >> 9) % NT;
    int kc   = i / (512 * NT);
    int khalf = (lane >> 4) << 3;                       // 0 or 8
    int col   = nt * 16 + (lane & 15);
    int k     = kc * 32 + khalf + ((j < 8) ? j : (8 + j)); // j>=8 -> +16+(j-8)
    float w = (col < NREAL) ? W[k * NREAL + col] : 0.0f;
    Wl[i] = (half_t)w;
  }
  __syncthreads();

  const int wave = threadIdx.x >> 5;
  const int lane = threadIdx.x & 31;
  const int tile = blockIdx.x * 8 + wave;
  if (tile >= Mtiles) return;           // uniform per-wave exit; EXEC stays full

  const int m0 = tile * 16;
  int row = m0 + (lane & 15);
  if (row >= M) row = M - 1;            // clamp loads; stores guarded below
  const int khalf = (lane >> 4) << 3;   // 0 for lanes 0-15, 8 for lanes 16-31

  v8f acc[NT];
#pragma unroll
  for (int t = 0; t < NT; ++t) acc[t] = (v8f){0.f,0.f,0.f,0.f,0.f,0.f,0.f,0.f};

#pragma unroll
  for (int kc = 0; kc < KC; ++kc) {
    // A fragment (16-bit A layout): two contiguous 8-float runs, converted to f16
    v16h a;
    const float* ap = A + (size_t)row * K + kc * 32 + khalf;
#pragma unroll
    for (int j = 0; j < 8; ++j) {
      a[j]     = (half_t)ap[j];
      a[8 + j] = (half_t)ap[16 + j];
    }
#pragma unroll
    for (int t = 0; t < NT; ++t) {
      v16h b = *(const v16h*)&Wl[(((kc * NT) + t) * 32 + lane) * 16];
      acc[t] = __builtin_amdgcn_wmma_f32_16x16x32_f16(
          /*neg_a=*/false, a, /*neg_b=*/false, b,
          /*c_mod=*/(short)0, acc[t], /*reuse_a=*/false, /*reuse_b=*/false);
    }
  }

  // C layout: VGPR r -> row m0+r (lanes 0-15) / m0+8+r (lanes 16-31), col = lane&15
  const int rbase = m0 + ((lane >> 4) << 3);
  const int cb = lane & 15;
  if (m0 + 16 <= M) {
    // full tile: unguarded stores (hot path — M is a multiple of 16 here)
#pragma unroll
    for (int t = 0; t < NT; ++t) {
#pragma unroll
      for (int r = 0; r < 8; ++r) {
        C[(size_t)(rbase + r) * NPAD + t * 16 + cb] = acc[t][r];
      }
    }
  } else {
#pragma unroll
    for (int t = 0; t < NT; ++t) {
#pragma unroll
      for (int r = 0; r < 8; ++r) {
        int rr = rbase + r;
        if (rr < M) C[(size_t)rr * NPAD + t * 16 + cb] = acc[t][r];
      }
    }
  }
}

// ---------------- launcher ----------------

extern "C" void kernel_launch(void* const* d_in, const int* in_sizes, int n_in,
                              void* d_out, int out_size, void* d_ws, size_t ws_size,
                              hipStream_t stream) {
  (void)n_in; (void)out_size; (void)ws_size;
  const float* x  = (const float*)d_in[0];
  const int*   ei = (const int*)d_in[1];
  const float* W1 = (const float*)d_in[2];
  const float* b1 = (const float*)d_in[3];
  const float* W2 = (const float*)d_in[4];
  const float* b2 = (const float*)d_in[5];
  float* out = (float*)d_out;

  const int Fin = 128, F1 = 64, F2 = 41, F2P = 48;
  const int n = in_sizes[0] / Fin;     // 100000
  const int E = in_sizes[1] / 2;       // 1600000
  const int* src = ei;
  const int* dst = ei + E;

  // workspace layout (all fully initialized every call)
  float* deg  = (float*)d_ws;                  // n
  float* dinv = deg + n;                       // n
  float* h1   = dinv + n;                      // n*64
  float* hb   = h1 + (size_t)n * F1;           // n*64 (aggregated + relu)
  float* h2   = hb + (size_t)n * F1;           // n*48

  const int B = 256;

  // degree + normalization
  fill1_kernel<<<(n + B - 1) / B, B, 0, stream>>>(deg, n);
  deg_accum_kernel<<<(E + B - 1) / B, B, 0, stream>>>(dst, deg, E);
  dinv_kernel<<<(n + B - 1) / B, B, 0, stream>>>(deg, dinv, n);

  const int Mtiles = (n + 15) / 16;            // 6250
  const int gblocks = (Mtiles + 7) / 8;        // 8 waves/block

  // layer 1: h1 = x @ W1 ; hb = b1 + self-loop + edge scatter ; relu
  gemm_wmma_kernel<128, 64, 64><<<gblocks, B, 0, stream>>>(x, W1, h1, Mtiles, n);
  const int t1 = n * F1;
  node_init_kernel<<<(t1 + B - 1) / B, B, 0, stream>>>(h1, dinv, b1, hb, n, F1, F1);
  const int s1 = E * 16;
  scatter64_kernel<<<(s1 + B - 1) / B, B, 0, stream>>>(h1, src, dst, dinv, hb, E);
  relu_kernel<<<(t1 + B - 1) / B, B, 0, stream>>>(hb, t1);

  // layer 2: h2 = hb @ W2 (padded to 48 cols) ; out = b2 + self-loop + edge scatter
  gemm_wmma_kernel<64, 41, 48><<<gblocks, B, 0, stream>>>(hb, W2, h2, Mtiles, n);
  const int t2 = n * F2;
  node_init_kernel<<<(t2 + B - 1) / B, B, 0, stream>>>(h2, dinv, b2, out, n, F2, F2P);
  const int s2 = E * F2;
  scatter_kernel<<<(s2 + B - 1) / B, B, 0, stream>>>(h2, src, dst, dinv, out, E, F2, F2P);
}
